// FourierBlock_21174188769471
// MI455X (gfx1250) — matrix operand order repaired and checked by reading, not compiled, not run on hardware
//
#include <hip/hip_runtime.h>

// FNO spectral block as 3 WMMA GEMMs. ~100 MB traffic -> HBM bound on MI455X.
// All matrix math on v_wmma_f32_16x16x32_f16 (f16 in, f32 accum).
// All LDS operand tiles stored so each lane's fragment is contiguous (ds_load_b128).

typedef __attribute__((ext_vector_type(16))) _Float16 v16h;
typedef __attribute__((ext_vector_type(8)))  float    v8f;

#define MODES 64
#define LEN   2048
#define BATCH 8
#define HEADS 8
#define EDIM  64
#define NCOL  128            // 2*MODES (real | imag columns)
#define KSPLIT 4             // forward DFT K-split for occupancy
#define WSCALE 8192.0f       // lift w (~4e-6) into f16 normal range
#define PI_F 3.14159265358979323846f

// ---- workspace layout (bytes) ----
#define OFF_BFWD 0                                       // f16 [n=128][l=2048] (transposed)
#define OFF_BINV (OFF_BFWD + LEN*NCOL*2)                 // f16 [l=2048][n=128] (transposed)
#define OFF_BW   (OFF_BINV + NCOL*LEN*2)                 // f16 [H][M][n=128][k=128]
#define OFF_XQP  (OFF_BW + HEADS*MODES*NCOL*NCOL*2)      // f32 [KSPLIT][H][M][B][128]
#define XQP_STRIDE (HEADS*MODES*BATCH*NCOL)              // floats per K-slice
#define OFF_Y    (OFF_XQP + KSPLIT*XQP_STRIDE*4)         // f16 [B][H][E][128]
// total ~27 MB

// ---------- WMMA fragment loads from LDS (contiguous k order, wave32) ----------
// A tile layout [row][k], ld mult of 8: lane%16 = M row;
// lanes<16 hold K 0..15, lanes>=16 hold K 16..31. 32B contiguous.
__device__ inline v16h frag_a(const _Float16* tile, int ld, int row_base, int lane) {
  const _Float16* p = tile + (row_base + (lane & 15)) * ld + ((lane < 16) ? 0 : 16);
  v16h a;
#pragma unroll
  for (int j = 0; j < 16; ++j) a[j] = p[j];
  return a;
}
// B stored TRANSPOSED in LDS: tile layout [n][k], ld mult of 8.
// lane%16 = N col; lanes<16 hold K 0..15, lanes>=16 hold K 16..31. 32B contiguous.
__device__ inline v16h frag_bt(const _Float16* tile, int ld, int col_base, int lane) {
  const _Float16* p = tile + (col_base + (lane & 15)) * ld + ((lane < 16) ? 0 : 16);
  v16h b;
#pragma unroll
  for (int j = 0; j < 16; ++j) b[j] = p[j];
  return b;
}

// ---------- kernel 0a: DFT twiddle tables (f16, transposed layouts) ----------
__global__ void k_tables(_Float16* __restrict__ bfwd_t, _Float16* __restrict__ binv_t) {
  int l = blockIdx.x;        // 0..2047
  int n = threadIdx.x;       // 0..127
  int m = n & 63;
  int phase = (m * l) & (LEN - 1);
  float ang = (2.0f * PI_F / (float)LEN) * (float)phase;
  float c = __cosf(ang), s = __sinf(ang);
  // forward: Xr[m] = sum_l x cos  ;  Xi[m] = -sum_l x sin      (stored [n][l])
  bfwd_t[n * LEN + l] = (_Float16)((n < 64) ? c : -s);
  // inverse: y[l] = (1/L)[Yr0 + 2*sum_{m>=1}(Yr cos - Yi sin)] (stored [l][n])
  float v;
  if (n == 0)       v = 1.0f / (float)LEN;
  else if (n < 64)  v = (2.0f / (float)LEN) * c;
  else if (n == 64) v = 0.0f;                    // imag of m=0 dropped by irfft
  else              v = -(2.0f / (float)LEN) * s;
  binv_t[l * NCOL + n] = (_Float16)v;
}

// ---------- kernel 0b: repack weights, per-(h,m) [[Wr,Wi],[-Wi,Wr]] stored [n][k] ----------
__global__ __launch_bounds__(256) void k_repack(const float* __restrict__ wr,
                                                const float* __restrict__ wi,
                                                _Float16* __restrict__ bw) {
  int h = blockIdx.x >> 6, m = blockIdx.x & 63;
  _Float16* dst = bw + (size_t)(h * MODES + m) * NCOL * NCOL;
  for (int idx = threadIdx.x; idx < NCOL * NCOL; idx += 256) {
    int n = idx >> 7, k = idx & 127;             // [n][k] layout
    int i = k & 63, o = n & 63;
    size_t src = (((size_t)h * EDIM + i) * EDIM + o) * MODES + m;
    float v;
    if (k < 64) v = (n < 64) ?  wr[src] : wi[src];
    else        v = (n < 64) ? -wi[src] : wr[src];
    dst[idx] = (_Float16)(v * WSCALE);
  }
}

// ---------- kernel 1: forward truncated DFT (K-split partials, f32 out) ----------
__global__ __launch_bounds__(256) void k_fwd(const float* __restrict__ x,
                                             const _Float16* __restrict__ bfwd_t,
                                             float* __restrict__ xqp) {
  __shared__ _Float16 As[64 * 32];     // [e][kk]
  __shared__ _Float16 Bt[128 * 32];    // [n][kk]  (transposed tile)
  const int tid = threadIdx.x, lane = tid & 31, wave = tid >> 5;
  const int b = blockIdx.x >> 3, h = blockIdx.x & 7;
  const int ks = blockIdx.y;                     // K slice
  const int mt = (wave >> 1) * 16;               // e tile
  const int nb = (wave & 1) * 64;                // n half
  v8f acc[4] = {};
  const int kbeg = ks * (LEN / KSPLIT), kend = kbeg + LEN / KSPLIT;
  for (int k0 = kbeg; k0 < kend; k0 += 32) {
#pragma unroll
    for (int p = 0; p < 8; ++p) {                // 64x32 A tile (x gathered, coalesced in e)
      int idx = p * 256 + tid;
      int e = idx & 63, kk = idx >> 6;
      As[e * 32 + kk] =
          (_Float16)x[(((size_t)b * LEN + (k0 + kk)) * HEADS + h) * EDIM + e];
    }
#pragma unroll
    for (int p = 0; p < 2; ++p) {                // 128x32 twiddle tile: 512 uint4 rows-of-4
      int idx = p * 256 + tid;
      int n = idx >> 2, q = idx & 3;
      ((uint4*)(Bt + n * 32))[q] = ((const uint4*)(bfwd_t + (size_t)n * LEN + k0))[q];
    }
    __syncthreads();
    v16h af = frag_a(As, 32, mt, lane);
#pragma unroll
    for (int nt = 0; nt < 4; ++nt) {
      v16h bf = frag_bt(Bt, 32, nb + nt * 16, lane);
      acc[nt] = __builtin_amdgcn_wmma_f32_16x16x32_f16(false, af, false, bf,
                                                       (short)0, acc[nt], false, false);
    }
    __syncthreads();
  }
  float* dst = xqp + (size_t)ks * XQP_STRIDE;    // [h][m][b][img*64+e] f32 partial
#pragma unroll
  for (int nt = 0; nt < 4; ++nt) {
    int n = nb + nt * 16 + (lane & 15);
    int m = n & 63, img = n >> 6;
#pragma unroll
    for (int r = 0; r < 8; ++r) {
      int e = mt + r + ((lane < 16) ? 0 : 8);
      dst[((h * MODES + m) * BATCH + b) * NCOL + img * 64 + e] = acc[nt][r];
    }
  }
}

// ---------- kernel 2: per-(h,m) complex mode mixing, 16x128x128 GEMM ----------
__global__ __launch_bounds__(256) void k_mix(const float* __restrict__ xqp,
                                             const _Float16* __restrict__ bw,
                                             _Float16* __restrict__ y) {
  __shared__ _Float16 As[16 * 128];    // [row=b(8 valid + 8 pad)][k]
  __shared__ _Float16 Ws[128 * 128];   // [n][k] (already transposed in global)
  const int tid = threadIdx.x, lane = tid & 31, wave = tid >> 5;
  const int h = blockIdx.x >> 6, m = blockIdx.x & 63;
  const size_t xoff = (size_t)((h * MODES + m) * BATCH) * NCOL;
  for (int idx = tid; idx < 16 * 128; idx += 256) {
    int row = idx >> 7, k = idx & 127;
    float v = 0.0f;
    if (row < 8) {
#pragma unroll
      for (int s = 0; s < KSPLIT; ++s)
        v += xqp[(size_t)s * XQP_STRIDE + xoff + row * NCOL + k];
    }
    As[idx] = (_Float16)v;
  }
  const uint4* wsrc = (const uint4*)(bw + (size_t)(h * MODES + m) * NCOL * NCOL);
  uint4* wdst = (uint4*)Ws;
  for (int idx = tid; idx < (NCOL * NCOL) / 8; idx += 256) wdst[idx] = wsrc[idx];
  __syncthreads();
  v8f acc = {};
#pragma unroll
  for (int kc = 0; kc < 4; ++kc) {
    v16h af = frag_a(As + kc * 32, 128, 0, lane);
    v16h bf = frag_bt(Ws + kc * 32, 128, wave * 16, lane);
    acc = __builtin_amdgcn_wmma_f32_16x16x32_f16(false, af, false, bf,
                                                 (short)0, acc, false, false);
  }
  if (lane < 16) {                               // rows M=0..7 (valid b) live in lanes<16
    int nout = wave * 16 + (lane & 15);
    int o = nout & 63, img = nout >> 6;
#pragma unroll
    for (int r = 0; r < 8; ++r)                  // b = r
      y[(((size_t)r * HEADS + h) * EDIM + o) * NCOL + img * 64 + m] =
          (_Float16)(acc[r] * (1.0f / WSCALE));
  }
}

// ---------- kernel 3: inverse truncated DFT + transpose epilogue ----------
__global__ __launch_bounds__(256) void k_inv(const _Float16* __restrict__ y,
                                             const _Float16* __restrict__ binv_t,
                                             float* __restrict__ out) {
  __shared__ _Float16 As[64 * 128];    // [o][n]  16 KB
  __shared__ _Float16 Bt[128 * 128];   // [l][n]  32 KB (transposed tile; aliased by Ds)
  const int tid = threadIdx.x, lane = tid & 31, wave = tid >> 5;
  const int l0 = blockIdx.x * 128;
  const int b = blockIdx.y >> 3, h = blockIdx.y & 7;
  {
    const uint4* asrc = (const uint4*)(y + ((size_t)b * HEADS + h) * EDIM * NCOL);
    uint4* adst = (uint4*)As;
    for (int idx = tid; idx < (64 * 128) / 8; idx += 256) adst[idx] = asrc[idx];
    for (int idx = tid; idx < 2048; idx += 256) {        // 128 l-rows x 16 uint4
      int row = idx >> 4, q = idx & 15;
      ((uint4*)(Bt + row * 128))[q] =
          ((const uint4*)(binv_t + (size_t)(l0 + row) * NCOL))[q];
    }
  }
  __syncthreads();
  const int mt = (wave >> 1) * 16, nb = (wave & 1) * 64;
  v8f acc[4] = {};
#pragma unroll
  for (int kc = 0; kc < 4; ++kc) {
    v16h af = frag_a(As + kc * 32, 128, mt, lane);
#pragma unroll
    for (int nt = 0; nt < 4; ++nt) {
      v16h bf = frag_bt(Bt + kc * 32, 128, nb + nt * 16, lane);
      acc[nt] = __builtin_amdgcn_wmma_f32_16x16x32_f16(false, af, false, bf,
                                                       (short)0, acc[nt], false, false);
    }
  }
  __syncthreads();                     // everyone done reading Bt -> reuse as Ds
  float* Ds = (float*)Bt;              // [l][e] 128*64 f32 = 32 KB
#pragma unroll
  for (int nt = 0; nt < 4; ++nt) {
    int l = nb + nt * 16 + (lane & 15);
#pragma unroll
    for (int r = 0; r < 8; ++r) {
      int e = mt + r + ((lane < 16) ? 0 : 8);
      Ds[l * 64 + e] = acc[nt][r];
    }
  }
  __syncthreads();
  for (int idx = tid; idx < 128 * 64; idx += 256) {      // coalesced (b,l,h,e) store
    int e = idx & 63, l = idx >> 6;
    out[(((size_t)b * LEN + (l0 + l)) * HEADS + h) * EDIM + e] = Ds[idx];
  }
}

extern "C" void kernel_launch(void* const* d_in, const int* in_sizes, int n_in,
                              void* d_out, int out_size, void* d_ws, size_t ws_size,
                              hipStream_t stream) {
  const float* x  = (const float*)d_in[0];
  const float* wr = (const float*)d_in[1];
  const float* wi = (const float*)d_in[2];
  float* out = (float*)d_out;
  char* ws = (char*)d_ws;
  _Float16* bfwd_t = (_Float16*)(ws + OFF_BFWD);
  _Float16* binv_t = (_Float16*)(ws + OFF_BINV);
  _Float16* bw     = (_Float16*)(ws + OFF_BW);
  float*    xqp    = (float*)   (ws + OFF_XQP);
  _Float16* yq     = (_Float16*)(ws + OFF_Y);

  k_tables<<<LEN, NCOL, 0, stream>>>(bfwd_t, binv_t);
  k_repack<<<HEADS * MODES, 256, 0, stream>>>(wr, wi, bw);
  k_fwd<<<dim3(BATCH * HEADS, KSPLIT), 256, 0, stream>>>(x, bfwd_t, xqp);
  k_mix<<<HEADS * MODES, 256, 0, stream>>>(xqp, bw, yq);
  k_inv<<<dim3(LEN / 128, BATCH * HEADS), 256, 0, stream>>>(yq, binv_t, out);
}
